// LinearLookbackAttention_7129645711714
// MI455X (gfx1250) — matrix operand order repaired and checked by reading, not compiled, not run on hardware
//
#include <hip/hip_runtime.h>

typedef __bf16 bf16;
typedef __bf16 v16bf __attribute__((ext_vector_type(16)));
typedef float  v8f   __attribute__((ext_vector_type(8)));

#define WMMA_BF16(A_, B_, C_) \
  __builtin_amdgcn_wmma_f32_16x16x32_bf16(false, (A_), false, (B_), (short)0, (C_), false, false)

// Problem dims
#define B_SZ   4
#define T_SZ   2048
#define DM_SZ  768
#define H_SZ   4
#define D_SZ   64
#define DIM_SZ 256
#define MROWS  (B_SZ * T_SZ)     // 8192
#define CHUNK  64
#define NCHUNK (T_SZ / CHUNK)    // 32
#define NCHAIN (B_SZ * H_SZ)     // 16
#define NQKVG  (3 * DIM_SZ + DM_SZ)  // 1536 fused projection columns

// ---------------------------------------------------------------------------
// Fragment load helpers (contiguous k order k = 16*hi + e, same in A and B)
// A 16x32 bf16: lane half hi selects K sub-block; element e -> k = hi*16 + e
__device__ __forceinline__ v16bf load_a_frag(const bf16* __restrict__ arow, int k0, int hi) {
  v16bf af;
#pragma unroll
  for (int e = 0; e < 8; ++e) {
    af[e]     = arow[k0 + hi * 16 + e];
    af[8 + e] = arow[k0 + hi * 16 + 8 + e];
  }
  return af;
}
// B 32x16 bf16 from [N][K] storage: lane's 16 contiguous K values
__device__ __forceinline__ v16bf load_b_frag(const bf16* __restrict__ bptr) {
  v16bf bf;
#pragma unroll
  for (int e = 0; e < 16; ++e) bf[e] = bptr[e];
  return bf;
}

// ---------------------------------------------------------------------------
__global__ void cvt_bf16_kernel(const float* __restrict__ in, bf16* __restrict__ out, int n) {
  int i = blockIdx.x * blockDim.x + threadIdx.x;
  if (i < n) out[i] = (bf16)in[i];
}

// in [K,N] f32 -> out [N,K] bf16 (K contiguous for B-fragments)
__global__ void transpose_w_kernel(const float* __restrict__ in, bf16* __restrict__ out,
                                   int K, int N) {
  int i = blockIdx.x * blockDim.x + threadIdx.x;
  if (i < K * N) {
    int n = i % N, k = i / N;
    out[(size_t)n * K + k] = (bf16)in[i];
  }
}

// ---------------------------------------------------------------------------
// Fused QKV+gate projection: [8192,768] x [1536,768]^T, double-buffered k-loop.
// Column routing: n<512 -> elu+1 (Q,K); n<768 -> identity (V); else sigmoid (gate).
__global__ __launch_bounds__(256) void proj_fused_kernel(
    const bf16* __restrict__ A, const bf16* __restrict__ Wt,
    bf16* __restrict__ Qb, bf16* __restrict__ Kb, bf16* __restrict__ Vb,
    bf16* __restrict__ gateb)
{
  const int lane = threadIdx.x & 31;
  const int wave = threadIdx.x >> 5;
  const int mBase = blockIdx.x * 128 + wave * 16;
  const int nBase = blockIdx.y * 64;
  const int nlow = lane & 15, hi = lane >> 4;
  const int K = DM_SZ;

  const bf16* arow = A + (size_t)(mBase + nlow) * K;
  const bf16* b0p = Wt + (size_t)(nBase +  0 + nlow) * K + hi * 16;
  const bf16* b1p = Wt + (size_t)(nBase + 16 + nlow) * K + hi * 16;
  const bf16* b2p = Wt + (size_t)(nBase + 32 + nlow) * K + hi * 16;
  const bf16* b3p = Wt + (size_t)(nBase + 48 + nlow) * K + hi * 16;

  v8f acc0 = {}, acc1 = {}, acc2 = {}, acc3 = {};
  v16bf af = load_a_frag(arow, 0, hi);
  v16bf b0 = load_b_frag(b0p), b1 = load_b_frag(b1p);
  v16bf b2 = load_b_frag(b2p), b3 = load_b_frag(b3p);

  for (int k0 = 32; k0 < K; k0 += 32) {
    __builtin_prefetch(arow + k0 + 128, 0, 1);   // global_prefetch_b8 ahead of A stream
    v16bf afn = load_a_frag(arow, k0, hi);
    v16bf b0n = load_b_frag(b0p + k0), b1n = load_b_frag(b1p + k0);
    v16bf b2n = load_b_frag(b2p + k0), b3n = load_b_frag(b3p + k0);
    acc0 = WMMA_BF16(af, b0, acc0);
    acc1 = WMMA_BF16(af, b1, acc1);
    acc2 = WMMA_BF16(af, b2, acc2);
    acc3 = WMMA_BF16(af, b3, acc3);
    af = afn; b0 = b0n; b1 = b1n; b2 = b2n; b3 = b3n;
  }
  acc0 = WMMA_BF16(af, b0, acc0);
  acc1 = WMMA_BF16(af, b1, acc1);
  acc2 = WMMA_BF16(af, b2, acc2);
  acc3 = WMMA_BF16(af, b3, acc3);

#pragma unroll
  for (int nt = 0; nt < 4; ++nt) {
    v8f acc = (nt == 0) ? acc0 : (nt == 1) ? acc1 : (nt == 2) ? acc2 : acc3;
    int n = nBase + nt * 16 + nlow;
#pragma unroll
    for (int i = 0; i < 8; ++i) {
      int m = mBase + i + hi * 8;
      float v = acc[i];
      if (n < 512)      v = (v > 0.f) ? (v + 1.f) : __expf(v);   // elu(x)+1
      else if (n >= 768) v = 1.f / (1.f + __expf(-v));           // sigmoid
      if (n < 256)      Qb[(size_t)m * DIM_SZ + n]          = (bf16)v;
      else if (n < 512) Kb[(size_t)m * DIM_SZ + (n - 256)]  = (bf16)v;
      else if (n < 768) Vb[(size_t)m * DIM_SZ + (n - 512)]  = (bf16)v;
      else              gateb[(size_t)m * DM_SZ + (n - 768)] = (bf16)v;
    }
  }
}

// ---------------------------------------------------------------------------
// Final projection: Y = (out_att[8192,256] @ Wo) * gate, f32 out, double-buffered.
__global__ __launch_bounds__(256) void final_gemm_kernel(
    const bf16* __restrict__ A, const bf16* __restrict__ Wt,
    const bf16* __restrict__ gate, float* __restrict__ Y)
{
  const int lane = threadIdx.x & 31;
  const int wave = threadIdx.x >> 5;
  const int mBase = blockIdx.x * 128 + wave * 16;
  const int nBase = blockIdx.y * 64;
  const int nlow = lane & 15, hi = lane >> 4;
  const int K = DIM_SZ;

  const bf16* arow = A + (size_t)(mBase + nlow) * K;
  const bf16* b0p = Wt + (size_t)(nBase +  0 + nlow) * K + hi * 16;
  const bf16* b1p = Wt + (size_t)(nBase + 16 + nlow) * K + hi * 16;
  const bf16* b2p = Wt + (size_t)(nBase + 32 + nlow) * K + hi * 16;
  const bf16* b3p = Wt + (size_t)(nBase + 48 + nlow) * K + hi * 16;

  v8f acc0 = {}, acc1 = {}, acc2 = {}, acc3 = {};
  v16bf af = load_a_frag(arow, 0, hi);
  v16bf b0 = load_b_frag(b0p), b1 = load_b_frag(b1p);
  v16bf b2 = load_b_frag(b2p), b3 = load_b_frag(b3p);

  for (int k0 = 32; k0 < K; k0 += 32) {
    v16bf afn = load_a_frag(arow, k0, hi);
    v16bf b0n = load_b_frag(b0p + k0), b1n = load_b_frag(b1p + k0);
    v16bf b2n = load_b_frag(b2p + k0), b3n = load_b_frag(b3p + k0);
    acc0 = WMMA_BF16(af, b0, acc0);
    acc1 = WMMA_BF16(af, b1, acc1);
    acc2 = WMMA_BF16(af, b2, acc2);
    acc3 = WMMA_BF16(af, b3, acc3);
    af = afn; b0 = b0n; b1 = b1n; b2 = b2n; b3 = b3n;
  }
  acc0 = WMMA_BF16(af, b0, acc0);
  acc1 = WMMA_BF16(af, b1, acc1);
  acc2 = WMMA_BF16(af, b2, acc2);
  acc3 = WMMA_BF16(af, b3, acc3);

#pragma unroll
  for (int nt = 0; nt < 4; ++nt) {
    v8f acc = (nt == 0) ? acc0 : (nt == 1) ? acc1 : (nt == 2) ? acc2 : acc3;
    int n = nBase + nt * 16 + nlow;
#pragma unroll
    for (int i = 0; i < 8; ++i) {
      int m = mBase + i + hi * 8;
      size_t idx = (size_t)m * DM_SZ + n;
      Y[idx] = acc[i] * (float)gate[idx];
    }
  }
}

// ---------------------------------------------------------------------------
// Per-chunk local stats: S_loc = K_c^T V_c  (64x64 f32), z_loc = colsum(K_c)
__global__ __launch_bounds__(128) void chunk_stats_kernel(
    const bf16* __restrict__ Kb, const bf16* __restrict__ Vb,
    float* __restrict__ S_loc, float* __restrict__ z_loc)
{
  __shared__ bf16 ldsK[64 * 72];
  __shared__ bf16 ldsV[64 * 72];
  const int blk = blockIdx.x;           // chain*32 + chunk
  const int chain = blk >> 5, chunk = blk & 31;
  const int b = chain >> 2, h = chain & 3;
  const int tid = threadIdx.x, lane = tid & 31, wave = tid >> 5;
  const int nlow = lane & 15, hi = lane >> 4;

  const size_t rowBase = ((size_t)b * T_SZ + (size_t)chunk * CHUNK) * DIM_SZ + h * D_SZ;
  for (int idx = tid; idx < 64 * 64; idx += 128) {
    int t = idx >> 6, d = idx & 63;
    ldsK[t * 72 + d] = Kb[rowBase + (size_t)t * DIM_SZ + d];
    ldsV[t * 72 + d] = Vb[rowBase + (size_t)t * DIM_SZ + d];
  }
  __syncthreads();

  const int mBase = wave * 16;          // d rows
  v8f acc0 = {}, acc1 = {}, acc2 = {}, acc3 = {};
  for (int k0 = 0; k0 < 64; k0 += 32) { // reduce over t
    v16bf af;                           // A[m=d][k=t] = K[t][d] (LDS transpose gather)
    int m = mBase + nlow;
#pragma unroll
    for (int e = 0; e < 8; ++e) {
      af[e]     = ldsK[(k0 + hi * 16 + e) * 72 + m];
      af[8 + e] = ldsK[(k0 + hi * 16 + 8 + e) * 72 + m];
    }
#pragma unroll
    for (int nt = 0; nt < 4; ++nt) {
      v16bf bf;
      int n = nt * 16 + nlow;
#pragma unroll
      for (int e = 0; e < 16; ++e) bf[e] = ldsV[(k0 + hi * 16 + e) * 72 + n];
      if      (nt == 0) acc0 = WMMA_BF16(af, bf, acc0);
      else if (nt == 1) acc1 = WMMA_BF16(af, bf, acc1);
      else if (nt == 2) acc2 = WMMA_BF16(af, bf, acc2);
      else              acc3 = WMMA_BF16(af, bf, acc3);
    }
  }
  float* Sout = S_loc + (size_t)blk * 64 * 64;
#pragma unroll
  for (int nt = 0; nt < 4; ++nt) {
    v8f acc = (nt == 0) ? acc0 : (nt == 1) ? acc1 : (nt == 2) ? acc2 : acc3;
#pragma unroll
    for (int i = 0; i < 8; ++i)
      Sout[(mBase + i + hi * 8) * 64 + nt * 16 + nlow] = acc[i];
  }

  if (tid < 64) {
    float s = 0.f;
    for (int t = 0; t < 64; ++t) s += (float)ldsK[t * 72 + tid];
    z_loc[(size_t)blk * 64 + tid] = s;
  }
}

// ---------------------------------------------------------------------------
// Exclusive prefix over chunks per chain (S and z)
__global__ __launch_bounds__(256) void prefix_kernel(
    const float* __restrict__ S_loc, const float* __restrict__ z_loc,
    float* __restrict__ S_prev, float* __restrict__ z_prev)
{
  const int chain = blockIdx.x;
  const int tid = threadIdx.x;
  float run[16];
#pragma unroll
  for (int i = 0; i < 16; ++i) run[i] = 0.f;
  float runz = 0.f;
  for (int c = 0; c < NCHUNK; ++c) {
    size_t base = ((size_t)chain * NCHUNK + c) * 4096;
#pragma unroll
    for (int i = 0; i < 16; ++i) {
      int idx = tid + i * 256;
      S_prev[base + idx] = run[i];
      run[i] += S_loc[base + idx];
    }
    if (tid < 64) {
      size_t zb = ((size_t)chain * NCHUNK + c) * 64 + tid;
      z_prev[zb] = runz;
      runz += z_loc[zb];
    }
  }
}

// ---------------------------------------------------------------------------
// Per-chunk output: out = (mask(QK^T) V + Q S_prev) / den
__global__ __launch_bounds__(128) void chunk_out_kernel(
    const bf16* __restrict__ Qb, const bf16* __restrict__ Kb, const bf16* __restrict__ Vb,
    const float* __restrict__ S_prev, const float* __restrict__ z_prev,
    bf16* __restrict__ out_att)
{
  __shared__ float ldsAtt[64 * 65];
  __shared__ bf16  ldsV[64 * 72];
  __shared__ bf16  ldsS[64 * 72];
  __shared__ float ldsZ[64];
  __shared__ float ldsDen[64];

  const int blk = blockIdx.x;
  const int chain = blk >> 5, chunk = blk & 31;
  const int b = chain >> 2, h = chain & 3;
  const int tid = threadIdx.x, lane = tid & 31, wave = tid >> 5;
  const int nlow = lane & 15, hi = lane >> 4;

  const size_t rowBase = ((size_t)b * T_SZ + (size_t)chunk * CHUNK) * DIM_SZ + h * D_SZ;
  const float* Sp = S_prev + (size_t)blk * 4096;
  for (int idx = tid; idx < 4096; idx += 128) {
    int r = idx >> 6, cc = idx & 63;
    ldsV[r * 72 + cc] = Vb[rowBase + (size_t)r * DIM_SZ + cc];
    ldsS[r * 72 + cc] = (bf16)Sp[idx];
  }
  if (tid < 64) ldsZ[tid] = z_prev[(size_t)blk * 64 + tid];
  __syncthreads();

  const int iBase = wave * 16;
  const bf16* qrow = Qb + rowBase + (size_t)(iBase + nlow) * DIM_SZ;

  // 1) att = Q K^T with causal mask (j <= i), stored f32 in LDS
#pragma unroll
  for (int jt = 0; jt < 4; ++jt) {
    v8f acc = {};
    const bf16* krow = Kb + rowBase + (size_t)(jt * 16 + nlow) * DIM_SZ;
    for (int k0 = 0; k0 < 64; k0 += 32) {   // reduce over d
      v16bf af = load_a_frag(qrow, k0, hi);
      v16bf bf = load_b_frag(krow + k0 + hi * 16);
      acc = WMMA_BF16(af, bf, acc);
    }
#pragma unroll
    for (int i = 0; i < 8; ++i) {
      int irow = iBase + i + hi * 8;
      int j = jt * 16 + nlow;
      ldsAtt[irow * 65 + j] = (j <= irow) ? acc[i] : 0.f;
    }
  }
  __syncthreads();

  // 2) den_i = max(rowsum(att) + q_i . z_prev, 1e-6)
  if (tid < 64) {
    float rs = 0.f;
    for (int j = 0; j < 64; ++j) rs += ldsAtt[tid * 65 + j];
    const bf16* qr = Qb + rowBase + (size_t)tid * DIM_SZ;
    float qz = 0.f;
    for (int d = 0; d < 64; ++d) qz += (float)qr[d] * ldsZ[d];
    float den = rs + qz;
    ldsDen[tid] = (den < 1e-6f) ? 1e-6f : den;
  }
  __syncthreads();

  // 3) out = (att @ V + Q @ S_prev) / den
#pragma unroll
  for (int dt = 0; dt < 4; ++dt) {
    v8f acc = {};
    for (int k0 = 0; k0 < 64; k0 += 32) {
      v16bf af, bf;
      int m = iBase + nlow;
#pragma unroll
      for (int e = 0; e < 8; ++e) {         // A = masked att (f32 LDS -> bf16)
        af[e]     = (bf16)ldsAtt[m * 65 + k0 + hi * 16 + e];
        af[8 + e] = (bf16)ldsAtt[m * 65 + k0 + hi * 16 + 8 + e];
      }
#pragma unroll
      for (int e = 0; e < 16; ++e)          // B = V (LDS gather, reduce over j)
        bf[e] = ldsV[(k0 + hi * 16 + e) * 72 + dt * 16 + nlow];
      acc = WMMA_BF16(af, bf, acc);

      v16bf af2 = load_a_frag(qrow, k0, hi); // A = Q (reduce over d)
      v16bf bf2;
#pragma unroll
      for (int e = 0; e < 16; ++e)          // B = S_prev (LDS gather)
        bf2[e] = ldsS[(k0 + hi * 16 + e) * 72 + dt * 16 + nlow];
      acc = WMMA_BF16(af2, bf2, acc);
    }
#pragma unroll
    for (int i = 0; i < 8; ++i) {
      int irow = iBase + i + hi * 8;
      out_att[rowBase + (size_t)irow * DIM_SZ + dt * 16 + nlow] =
          (bf16)(acc[i] / ldsDen[irow]);
    }
  }
}

// ---------------------------------------------------------------------------
extern "C" void kernel_launch(void* const* d_in, const int* in_sizes, int n_in,
                              void* d_out, int out_size, void* d_ws, size_t ws_size,
                              hipStream_t stream) {
  const float* x  = (const float*)d_in[0];
  const float* Wq = (const float*)d_in[1];
  const float* Wk = (const float*)d_in[2];
  const float* Wv = (const float*)d_in[3];
  const float* Wo = (const float*)d_in[4];
  const float* Wg = (const float*)d_in[5];
  float* Y = (float*)d_out;

  char* ws = (char*)d_ws;
  size_t off = 0;
  auto carve = [&](size_t bytes) -> char* {
    char* p = ws + off;
    off += (bytes + 255) & ~(size_t)255;
    return p;
  };

  bf16* xb      = (bf16*)carve((size_t)MROWS * DM_SZ * 2);
  bf16* wqkvg_t = (bf16*)carve((size_t)NQKVG * DM_SZ * 2);   // [1536][768]
  bf16* wo_t    = (bf16*)carve((size_t)DM_SZ * DIM_SZ * 2);
  bf16* Qb      = (bf16*)carve((size_t)MROWS * DIM_SZ * 2);
  bf16* Kb      = (bf16*)carve((size_t)MROWS * DIM_SZ * 2);
  bf16* Vb      = (bf16*)carve((size_t)MROWS * DIM_SZ * 2);
  bf16* gateb   = (bf16*)carve((size_t)MROWS * DM_SZ * 2);
  float* S_loc  = (float*)carve((size_t)NCHAIN * NCHUNK * 4096 * 4);
  float* z_loc  = (float*)carve((size_t)NCHAIN * NCHUNK * 64 * 4);
  float* S_prev = (float*)carve((size_t)NCHAIN * NCHUNK * 4096 * 4);
  float* z_prev = (float*)carve((size_t)NCHAIN * NCHUNK * 64 * 4);
  bf16* out_att = (bf16*)carve((size_t)MROWS * DIM_SZ * 2);
  (void)ws_size; (void)in_sizes; (void)n_in; (void)out_size;

  // 0) convert x; transpose weights into fused [1536][768] buffer + Wo^T
  {
    int n = MROWS * DM_SZ;
    cvt_bf16_kernel<<<(n + 255) / 256, 256, 0, stream>>>(x, xb, n);
  }
  transpose_w_kernel<<<(DM_SZ * DIM_SZ + 255) / 256, 256, 0, stream>>>(
      Wq, wqkvg_t, DM_SZ, DIM_SZ);                              // rows 0..255
  transpose_w_kernel<<<(DM_SZ * DIM_SZ + 255) / 256, 256, 0, stream>>>(
      Wk, wqkvg_t + (size_t)DIM_SZ * DM_SZ, DM_SZ, DIM_SZ);     // rows 256..511
  transpose_w_kernel<<<(DM_SZ * DIM_SZ + 255) / 256, 256, 0, stream>>>(
      Wv, wqkvg_t + (size_t)2 * DIM_SZ * DM_SZ, DM_SZ, DIM_SZ); // rows 512..767
  transpose_w_kernel<<<(DM_SZ * DM_SZ + 255) / 256, 256, 0, stream>>>(
      Wg, wqkvg_t + (size_t)3 * DIM_SZ * DM_SZ, DM_SZ, DM_SZ);  // rows 768..1535
  transpose_w_kernel<<<(DIM_SZ * DM_SZ + 255) / 256, 256, 0, stream>>>(
      Wo, wo_t, DIM_SZ, DM_SZ);

  // 1) fused Q/K/V/gate projection (reads x once)
  dim3 blk(256);
  dim3 gProj(MROWS / 128, NQKVG / 64);    // 64 x 24
  proj_fused_kernel<<<gProj, blk, 0, stream>>>(xb, wqkvg_t, Qb, Kb, Vb, gateb);

  // 2) chunked linear-attention scan
  chunk_stats_kernel<<<NCHAIN * NCHUNK, 128, 0, stream>>>(Kb, Vb, S_loc, z_loc);
  prefix_kernel<<<NCHAIN, 256, 0, stream>>>(S_loc, z_loc, S_prev, z_prev);
  chunk_out_kernel<<<NCHAIN * NCHUNK, 128, 0, stream>>>(Qb, Kb, Vb, S_prev, z_prev, out_att);

  // 3) final projection with gate
  dim3 gFinal(MROWS / 128, DM_SZ / 64);   // 64 x 12
  final_gemm_kernel<<<gFinal, blk, 0, stream>>>(out_att, wo_t, gateb, Y);
}
